// NonLocalBlock_29429115912500
// MI455X (gfx1250) — matrix operand order repaired and checked by reading, not compiled, not run on hardware
//
#include <hip/hip_runtime.h>
#include <hip/hip_bf16.h>
#include <cstdint>

// ---------------------------------------------------------------------------
// NonLocalBlock (selfImage=True) for MI455X / gfx1250, wave32 + WMMA bf16
// + TDM (tensor_load_to_lds) double-buffered key tiles in the flash kernel.
// B=2, C=512, H=W=64 -> N=4096, Ci=256.
// ---------------------------------------------------------------------------

typedef __bf16 bf16;
typedef __attribute__((ext_vector_type(16))) __bf16 bf16x16;
typedef __attribute__((ext_vector_type(8)))  __bf16 bf16x8;
typedef __attribute__((ext_vector_type(8)))  float  f32x8;
typedef __attribute__((ext_vector_type(4)))  unsigned int u32x4;
typedef __attribute__((ext_vector_type(8)))  int i32x8;
typedef __attribute__((ext_vector_type(4)))  int i32x4;

#define DI __device__ __forceinline__

constexpr int B_  = 2;
constexpr int C_  = 512;
constexpr int Ci_ = 256;
constexpr int N_  = 4096;

// ---- workspace layout (bytes) ---------------------------------------------
constexpr size_t SZ_SNT = (size_t)B_ * N_ * C_ * 2;    // bf16 [B][N][C]
constexpr size_t SZ_W   = (size_t)Ci_ * C_ * 2;        // bf16 [O][C]
constexpr size_t SZ_PRJ = (size_t)B_ * N_ * Ci_ * 2;   // bf16 [B][N][Ci] / [B][Ci][N]

constexpr size_t OFF_SNT  = 0;
constexpr size_t OFF_MNT  = OFF_SNT + SZ_SNT;
constexpr size_t OFF_WRG  = OFF_MNT + SZ_SNT;
constexpr size_t OFF_WRT  = OFF_WRG + SZ_W;
constexpr size_t OFF_WRP  = OFF_WRT + SZ_W;
constexpr size_t OFF_WFT  = OFF_WRP + SZ_W;
constexpr size_t OFF_WFP  = OFF_WFT + SZ_W;
constexpr size_t OFF_WRW  = OFF_WFP + SZ_W;
constexpr size_t OFF_G    = OFF_WRW + SZ_W;    // [B][Ci][N]
constexpr size_t OFF_TH   = OFF_G   + SZ_PRJ;  // [B][N][Ci]
constexpr size_t OFF_PH   = OFF_TH  + SZ_PRJ;
constexpr size_t OFF_STH  = OFF_PH  + SZ_PRJ;
constexpr size_t OFF_SPH  = OFF_STH + SZ_PRJ;
constexpr size_t OFF_YBF  = OFF_SPH + SZ_PRJ;

// ---- WMMA fragment loaders (contiguous K order, same for A and B) ------------------
// A 16x32 bf16: lane L -> row L%16; 16 contiguous K at (L/16)*16.
DI bf16x16 load_a(const bf16* M, int ld, int lane, int row0, int k0) {
  const bf16* p = M + (size_t)(row0 + (lane & 15)) * ld + k0 + ((lane >> 4) << 4);
  bf16x8 lo = *(const bf16x8*)p;
  bf16x8 hi = *(const bf16x8*)(p + 8);
  return __builtin_shufflevector(lo, hi, 0,1,2,3,4,5,6,7,8,9,10,11,12,13,14,15);
}
// B 32x16 bf16: lane L -> col L%16; 16 contiguous K at (L/16)*16.
DI bf16x16 load_b(const bf16* M, int ld, int lane, int col0, int k0) {
  const bf16* p = M + (size_t)(col0 + (lane & 15)) * ld + k0 + ((lane >> 4) << 4);
  return *(const bf16x16*)p;
}

DI f32x8 wmma_bf16(bf16x16 a, bf16x16 b, f32x8 c) {
  return __builtin_amdgcn_wmma_f32_16x16x32_bf16(false, a, false, b, (short)0, c,
                                                 false, false);
}

DI f32x8 zero8() { f32x8 z = {0.f,0.f,0.f,0.f,0.f,0.f,0.f,0.f}; return z; }

// ---- TDM: async 2D tile load, global -> LDS (CDNA5 ISA ch. 8 D#) -----------
// tile_d0 = contiguous elements per row, tile_d1 = rows, d0_stride = row pitch
// (elements). tensor dims == tile dims so no OOB clipping. bf16 (2-byte) data.
DI void tdm_load_2d(unsigned lds_byte_off, const bf16* gptr,
                    unsigned tile_d0, unsigned tile_d1, unsigned d0_stride) {
  unsigned long long ga = (unsigned long long)(uintptr_t)gptr;
  u32x4 g0;
  g0[0] = 1u;                                   // count=1, user descriptor
  g0[1] = lds_byte_off;                         // lds_addr
  g0[2] = (unsigned)(ga & 0xffffffffu);         // global_addr[31:0]
  g0[3] = (unsigned)((ga >> 32) & 0x1ffffffu)   // global_addr[56:32]
        | (2u << 30);                           // type = 2 ("image")
  i32x8 g1;
  g1[0] = (int)(1u << 16);                      // data_size = 1 (2 bytes)
  g1[1] = (int)((tile_d0 & 0xffffu) << 16);                    // tensor_dim0 lo
  g1[2] = (int)((tile_d0 >> 16) | ((tile_d1 & 0xffffu) << 16));// d0 hi | d1 lo
  g1[3] = (int)((tile_d1 >> 16) | (tile_d0 << 16));            // d1 hi | tile_dim0
  g1[4] = (int)(tile_d1 & 0xffffu);                            // tile_dim1 (2D)
  g1[5] = (int)d0_stride;                                      // dim0 stride lo
  g1[6] = 0;
  g1[7] = 0;
  i32x4 z4 = {0, 0, 0, 0};
  i32x8 z8 = {0, 0, 0, 0, 0, 0, 0, 0};
  // 6-arg form (clang-23 / amdgpu-toolchain): g0, g1, g2, g3, extra, cpol
  __builtin_amdgcn_tensor_load_to_lds(g0, g1, z4, z4, z8, 0);
}

// ---------------------------------------------------------------------------
// 1) BN(eval)+ReLU, fp32 [B][C][N] -> bf16 [B][N][C] (LDS-tiled transpose)
// ---------------------------------------------------------------------------
__global__ void bn_relu_t_kernel(const float* __restrict__ x,
                                 const float* __restrict__ gamma,
                                 const float* __restrict__ beta,
                                 const float* __restrict__ mean,
                                 const float* __restrict__ var,
                                 bf16* __restrict__ out) {
  __shared__ float tile[16][17];
  const int b  = blockIdx.z;
  const int n0 = blockIdx.x * 16;
  const int c0 = blockIdx.y * 16;
  const int tx = threadIdx.x, ty = threadIdx.y;
  const int c  = c0 + ty;
  const float inv = gamma[c] * rsqrtf(var[c] + 1e-5f);
  const float sh  = beta[c] - mean[c] * inv;
  float v = x[(size_t)b * C_ * N_ + (size_t)c * N_ + n0 + tx];
  tile[ty][tx] = fmaxf(v * inv + sh, 0.f);
  __syncthreads();
  out[(size_t)b * N_ * C_ + (size_t)(n0 + ty) * C_ + c0 + tx] = (bf16)tile[tx][ty];
}

// ---------------------------------------------------------------------------
// 2) weight fp32 -> bf16
// ---------------------------------------------------------------------------
__global__ void w2bf_kernel(const float* __restrict__ src, bf16* __restrict__ dst,
                            int n) {
  int i = blockIdx.x * blockDim.x + threadIdx.x;
  if (i < n) dst[i] = (bf16)src[i];
}

// ---------------------------------------------------------------------------
// 3a) projection -> [B][N][Ci]  (out[n,o] = sum_c X[n,c]*W[o,c] + b[o])
//     software-pipelined: next K-step fragments load while current WMMAs run.
// ---------------------------------------------------------------------------
__global__ void __launch_bounds__(128)
proj_nt_kernel(const bf16* __restrict__ X,   // [B][N][C]
               const bf16* __restrict__ W,   // [Ci][C]
               const float* __restrict__ bias,
               bf16* __restrict__ out) {     // [B][N][Ci]
  const int lane = threadIdx.x & 31;
  const int wave = threadIdx.x >> 5;
  const int n0   = blockIdx.x * 16;
  const int b    = blockIdx.y;
  const int o0   = wave * 64;
  const bf16* Xb = X + (size_t)b * N_ * C_;
  f32x8 acc[4];
#pragma unroll
  for (int f = 0; f < 4; ++f) acc[f] = zero8();

  bf16x16 a = load_a(Xb, C_, lane, n0, 0);
  bf16x16 bb[4];
#pragma unroll
  for (int f = 0; f < 4; ++f) bb[f] = load_b(W, C_, lane, o0 + f * 16, 0);

#pragma unroll 2
  for (int k0 = 0; k0 < C_; k0 += 32) {
    const int kn = (k0 + 32 < C_) ? k0 + 32 : 0;
    bf16x16 an = load_a(Xb, C_, lane, n0, kn);
    bf16x16 bn[4];
#pragma unroll
    for (int f = 0; f < 4; ++f) bn[f] = load_b(W, C_, lane, o0 + f * 16, kn);
#pragma unroll
    for (int f = 0; f < 4; ++f) acc[f] = wmma_bf16(a, bb[f], acc[f]);
    a = an;
#pragma unroll
    for (int f = 0; f < 4; ++f) bb[f] = bn[f];
  }

  const int rbase = n0 + ((lane >> 4) << 3);
#pragma unroll
  for (int f = 0; f < 4; ++f) {
    const int col = o0 + f * 16 + (lane & 15);
    const float bv = bias[col];
#pragma unroll
    for (int r = 0; r < 8; ++r)
      out[(size_t)b * N_ * Ci_ + (size_t)(rbase + r) * Ci_ + col] =
          (bf16)(acc[f][r] + bv);
  }
}

// ---------------------------------------------------------------------------
// 3b) projection -> [B][Ci][N]  (out[o,n] = sum_c W[o,c]*X[n,c] + b[o])
// ---------------------------------------------------------------------------
__global__ void __launch_bounds__(128)
proj_on_kernel(const bf16* __restrict__ X,   // [B][N][C]
               const bf16* __restrict__ W,   // [Ci][C]
               const float* __restrict__ bias,
               bf16* __restrict__ out) {     // [B][Ci][N]
  const int lane = threadIdx.x & 31;
  const int wave = threadIdx.x >> 5;
  const int o0   = blockIdx.x * 16;
  const int nb   = blockIdx.y * 256 + wave * 64;
  const int b    = blockIdx.z;
  const bf16* Xb = X + (size_t)b * N_ * C_;
  f32x8 acc[4];
#pragma unroll
  for (int f = 0; f < 4; ++f) acc[f] = zero8();

  bf16x16 a = load_a(W, C_, lane, o0, 0);
  bf16x16 bb[4];
#pragma unroll
  for (int f = 0; f < 4; ++f) bb[f] = load_b(Xb, C_, lane, nb + f * 16, 0);

#pragma unroll 2
  for (int k0 = 0; k0 < C_; k0 += 32) {
    const int kn = (k0 + 32 < C_) ? k0 + 32 : 0;
    bf16x16 an = load_a(W, C_, lane, o0, kn);
    bf16x16 bn[4];
#pragma unroll
    for (int f = 0; f < 4; ++f) bn[f] = load_b(Xb, C_, lane, nb + f * 16, kn);
#pragma unroll
    for (int f = 0; f < 4; ++f) acc[f] = wmma_bf16(a, bb[f], acc[f]);
    a = an;
#pragma unroll
    for (int f = 0; f < 4; ++f) bb[f] = bn[f];
  }

#pragma unroll
  for (int r = 0; r < 8; ++r) {
    const int rowo = o0 + ((lane >> 4) << 3) + r;
    const float bv = bias[rowo];
#pragma unroll
    for (int f = 0; f < 4; ++f)
      out[(size_t)b * Ci_ * N_ + (size_t)rowo * N_ + nb + f * 16 + (lane & 15)] =
          (bf16)(acc[f][r] + bv);
  }
}

// ---------------------------------------------------------------------------
// 4) fused flash attention: softmax(alpha*(Th·Ph^T) + STh·SPh^T) · G
//    All 4 waves share key tiles: TDM double-buffers phi/selfphi/G into LDS.
//    LDS map (single __shared__ block at offset 0), element offsets:
//      phi  buf0/1 :     0 /  8192      (32 keys x 256 c)
//      sphi buf0/1 : 16384 / 24576
//      G    buf0/1 : 32768 / 40960      (256 o x 32 keys)
//      P staging   : 49152 + wave*512   (16 q x 32 keys)
// ---------------------------------------------------------------------------
__global__ void __launch_bounds__(128)
flash_attn_kernel(const bf16* __restrict__ Th,   // [B][N][Ci]
                  const bf16* __restrict__ Ph,   // [B][N][Ci]
                  const bf16* __restrict__ STh,  // [B][N][Ci]
                  const bf16* __restrict__ SPh,  // [B][N][Ci]
                  const bf16* __restrict__ G,    // [B][Ci][N]
                  const float* __restrict__ alpha,
                  bf16* __restrict__ Y) {        // [B][N][Ci]
  __shared__ __align__(128) bf16 smem[49152 + 2048];  // 100 KB

  const int lane = threadIdx.x & 31;
  const int wave = threadIdx.x >> 5;
  const int b    = blockIdx.y;
  const int q0   = (blockIdx.x * 4 + wave) * 16;
  const float alp = alpha[0];

  const bf16* Thb  = Th  + (size_t)b * N_ * Ci_;
  const bf16* Phb  = Ph  + (size_t)b * N_ * Ci_;
  const bf16* SThb = STh + (size_t)b * N_ * Ci_;
  const bf16* SPhb = SPh + (size_t)b * N_ * Ci_;
  const bf16* Gb   = G   + (size_t)b * Ci_ * N_;

  // persistent query A-fragments: 16 queries x Ci=256 (8 K-frags each)
  bf16x16 afT[8], afS[8];
#pragma unroll
  for (int f = 0; f < 8; ++f) {
    afT[f] = load_a(Thb,  Ci_, lane, q0, f * 32);
    afS[f] = load_a(SThb, Ci_, lane, q0, f * 32);
  }

  // prologue: TDM-fill buffer 0 (wave 0 issues; EXEC is ignored by TDM)
  if (wave == 0) {
    tdm_load_2d(0u * 2u,     Phb,  Ci_, 32, Ci_);
    tdm_load_2d(16384u * 2u, SPhb, Ci_, 32, Ci_);
    tdm_load_2d(32768u * 2u, Gb,   32, Ci_, N_);
  }
  __builtin_amdgcn_s_wait_tensorcnt(0);
  __syncthreads();

  f32x8 yacc[16];
#pragma unroll
  for (int of = 0; of < 16; ++of) yacc[of] = zero8();
  float mrow[8], lrow[8];
#pragma unroll
  for (int r = 0; r < 8; ++r) { mrow[r] = -__builtin_inff(); lrow[r] = 0.f; }

  for (int kb = 0; kb < N_; kb += 32) {
    const int cur = (kb >> 5) & 1;
    // async-prefetch next key tiles into the other buffer
    if (wave == 0 && kb + 32 < N_) {
      const unsigned nb = (unsigned)(cur ^ 1) * 8192u;
      tdm_load_2d(nb * 2u,             Phb  + (size_t)(kb + 32) * Ci_, Ci_, 32, Ci_);
      tdm_load_2d((16384u + nb) * 2u,  SPhb + (size_t)(kb + 32) * Ci_, Ci_, 32, Ci_);
      tdm_load_2d((32768u + nb) * 2u,  Gb + kb + 32,                   32, Ci_, N_);
    }
    const bf16* ph  = smem + cur * 8192;
    const bf16* sph = smem + 16384 + cur * 8192;
    const bf16* gg  = smem + 32768 + cur * 8192;

    // two 16x16 score tiles (keys kb..+15, kb+16..+31) from LDS
    f32x8 s0 = zero8(), s1 = zero8(), t0 = zero8(), t1 = zero8();
    const int bcol = (lane & 15) * 256 + ((lane >> 4) << 4);
#pragma unroll
    for (int f = 0; f < 8; ++f) {
      bf16x16 b0 = *(const bf16x16*)(ph + bcol + f * 32);
      bf16x16 b1 = *(const bf16x16*)(ph + bcol + 16 * 256 + f * 32);
      s0 = wmma_bf16(afT[f], b0, s0);
      s1 = wmma_bf16(afT[f], b1, s1);
      bf16x16 c0 = *(const bf16x16*)(sph + bcol + f * 32);
      bf16x16 c1 = *(const bf16x16*)(sph + bcol + 16 * 256 + f * 32);
      t0 = wmma_bf16(afS[f], c0, t0);
      t1 = wmma_bf16(afS[f], c1, t1);
    }
    // combined scores + online softmax (row = r + 8*(lane/16), col = lane%16)
    bf16* Plw = smem + 49152 + wave * 512;
#pragma unroll
    for (int r = 0; r < 8; ++r) {
      float v0 = alp * s0[r] + t0[r];
      float v1 = alp * s1[r] + t1[r];
      float v = fmaxf(v0, v1);
      v = fmaxf(v, __shfl_xor(v, 1, 32));
      v = fmaxf(v, __shfl_xor(v, 2, 32));
      v = fmaxf(v, __shfl_xor(v, 4, 32));
      v = fmaxf(v, __shfl_xor(v, 8, 32));
      const float mnew = fmaxf(mrow[r], v);
      const float sc = __expf(mrow[r] - mnew);
      mrow[r] = mnew;
      const float p0 = __expf(v0 - mnew);
      const float p1 = __expf(v1 - mnew);
      float ps = p0 + p1;
      ps += __shfl_xor(ps, 1, 32);
      ps += __shfl_xor(ps, 2, 32);
      ps += __shfl_xor(ps, 4, 32);
      ps += __shfl_xor(ps, 8, 32);
      lrow[r] = lrow[r] * sc + ps;
#pragma unroll
      for (int of = 0; of < 16; ++of) yacc[of][r] *= sc;
      const int prow = r + ((lane >> 4) << 3);
      Plw[prow * 32 + (lane & 15)]      = (bf16)p0;
      Plw[prow * 32 + 16 + (lane & 15)] = (bf16)p1;
    }
    asm volatile("s_wait_dscnt 0" ::: "memory");
    // reload P as an A 16x32 fragment (same wave, no barrier needed)
    const bf16* pp = Plw + (lane & 15) * 32 + ((lane >> 4) << 4);
    bf16x8 plo = *(const bf16x8*)pp;
    bf16x8 phi = *(const bf16x8*)(pp + 8);
    bf16x16 pa = __builtin_shufflevector(plo, phi,
                                         0,1,2,3,4,5,6,7,8,9,10,11,12,13,14,15);
    // y += P · G  from LDS (col = output channel, K = keys)
#pragma unroll
    for (int of = 0; of < 16; ++of) {
      bf16x16 gbf = *(const bf16x16*)(gg + (of * 16 + (lane & 15)) * 32 +
                                      ((lane >> 4) << 4));
      yacc[of] = wmma_bf16(pa, gbf, yacc[of]);
    }
    // next buffer ready + everyone done with current before it is overwritten
    __builtin_amdgcn_s_wait_tensorcnt(0);
    __syncthreads();
  }

  // epilogue: divide by row sums, write bf16 [B][N][Ci]
#pragma unroll
  for (int of = 0; of < 16; ++of) {
    const int col = of * 16 + (lane & 15);
#pragma unroll
    for (int r = 0; r < 8; ++r) {
      const int row = q0 + r + ((lane >> 4) << 3);
      Y[(size_t)b * N_ * Ci_ + (size_t)row * Ci_ + col] =
          (bf16)(yacc[of][r] / lrow[r]);
    }
  }
}

// ---------------------------------------------------------------------------
// 5) out[c,n] = sum_o RW[c,o]*y[n,o] + RW_b[c] + self_fea[c,n]   (fp32 out)
// ---------------------------------------------------------------------------
__global__ void __launch_bounds__(128)
rw_residual_kernel(const bf16* __restrict__ Ybf,  // [B][N][Ci]
                   const bf16* __restrict__ W,    // [C][Ci]
                   const float* __restrict__ bias,
                   const float* __restrict__ resid, // [B][C][N]
                   float* __restrict__ out) {       // [B][C][N]
  const int lane = threadIdx.x & 31;
  const int wave = threadIdx.x >> 5;
  const int c0   = blockIdx.x * 16;
  const int nb   = blockIdx.y * 256 + wave * 64;
  const int b    = blockIdx.z;
  const bf16* Yb = Ybf + (size_t)b * N_ * Ci_;
  f32x8 acc[4];
#pragma unroll
  for (int f = 0; f < 4; ++f) acc[f] = zero8();

  bf16x16 a = load_a(W, Ci_, lane, c0, 0);
  bf16x16 bb[4];
#pragma unroll
  for (int f = 0; f < 4; ++f) bb[f] = load_b(Yb, Ci_, lane, nb + f * 16, 0);

#pragma unroll 2
  for (int k0 = 0; k0 < Ci_; k0 += 32) {
    const int kn = (k0 + 32 < Ci_) ? k0 + 32 : 0;
    bf16x16 an = load_a(W, Ci_, lane, c0, kn);
    bf16x16 bn[4];
#pragma unroll
    for (int f = 0; f < 4; ++f) bn[f] = load_b(Yb, Ci_, lane, nb + f * 16, kn);
#pragma unroll
    for (int f = 0; f < 4; ++f) acc[f] = wmma_bf16(a, bb[f], acc[f]);
    a = an;
#pragma unroll
    for (int f = 0; f < 4; ++f) bb[f] = bn[f];
  }

#pragma unroll
  for (int r = 0; r < 8; ++r) {
    const int rowc = c0 + ((lane >> 4) << 3) + r;
    const float bv = bias[rowc];
#pragma unroll
    for (int f = 0; f < 4; ++f) {
      const size_t idx =
          (size_t)b * C_ * N_ + (size_t)rowc * N_ + nb + f * 16 + (lane & 15);
      out[idx] = acc[f][r] + bv + resid[idx];
    }
  }
}

// ---------------------------------------------------------------------------
extern "C" void kernel_launch(void* const* d_in, const int* in_sizes, int n_in,
                              void* d_out, int out_size, void* d_ws, size_t ws_size,
                              hipStream_t stream) {
  const float* self_fea   = (const float*)d_in[0];
  const float* mutual_fea = (const float*)d_in[1];
  const float* alpha      = (const float*)d_in[2];
  const float* bn_i_gamma = (const float*)d_in[3];
  const float* bn_i_beta  = (const float*)d_in[4];
  const float* bn_i_mean  = (const float*)d_in[5];
  const float* bn_i_var   = (const float*)d_in[6];
  const float* bn_d_gamma = (const float*)d_in[7];
  const float* bn_d_beta  = (const float*)d_in[8];
  const float* bn_d_mean  = (const float*)d_in[9];
  const float* bn_d_var   = (const float*)d_in[10];
  const float* Rg_w = (const float*)d_in[11];
  const float* Rg_b = (const float*)d_in[12];
  const float* Rt_w = (const float*)d_in[13];
  const float* Rt_b = (const float*)d_in[14];
  const float* Rp_w = (const float*)d_in[15];
  const float* Rp_b = (const float*)d_in[16];
  const float* RW_w = (const float*)d_in[17];
  const float* RW_b = (const float*)d_in[18];
  const float* Ft_w = (const float*)d_in[19];
  const float* Ft_b = (const float*)d_in[20];
  const float* Fp_w = (const float*)d_in[21];
  const float* Fp_b = (const float*)d_in[22];

  char* ws = (char*)d_ws;
  bf16* snT = (bf16*)(ws + OFF_SNT);
  bf16* mnT = (bf16*)(ws + OFF_MNT);
  bf16* wRg = (bf16*)(ws + OFF_WRG);
  bf16* wRt = (bf16*)(ws + OFF_WRT);
  bf16* wRp = (bf16*)(ws + OFF_WRP);
  bf16* wFt = (bf16*)(ws + OFF_WFT);
  bf16* wFp = (bf16*)(ws + OFF_WFP);
  bf16* wRW = (bf16*)(ws + OFF_WRW);
  bf16* g   = (bf16*)(ws + OFF_G);
  bf16* th  = (bf16*)(ws + OFF_TH);
  bf16* ph  = (bf16*)(ws + OFF_PH);
  bf16* sth = (bf16*)(ws + OFF_STH);
  bf16* sph = (bf16*)(ws + OFF_SPH);
  bf16* ybf = (bf16*)(ws + OFF_YBF);

  // 1) BN+ReLU + transpose to bf16 [B][N][C]
  dim3 gT(N_ / 16, C_ / 16, B_), bT(16, 16);
  bn_relu_t_kernel<<<gT, bT, 0, stream>>>(self_fea, bn_i_gamma, bn_i_beta,
                                          bn_i_mean, bn_i_var, snT);
  bn_relu_t_kernel<<<gT, bT, 0, stream>>>(mutual_fea, bn_d_gamma, bn_d_beta,
                                          bn_d_mean, bn_d_var, mnT);

  // 2) weights -> bf16
  const int wn = Ci_ * C_;
  const int wb = (wn + 255) / 256;
  w2bf_kernel<<<wb, 256, 0, stream>>>(Rg_w, wRg, wn);
  w2bf_kernel<<<wb, 256, 0, stream>>>(Rt_w, wRt, wn);
  w2bf_kernel<<<wb, 256, 0, stream>>>(Rp_w, wRp, wn);
  w2bf_kernel<<<wb, 256, 0, stream>>>(Ft_w, wFt, wn);
  w2bf_kernel<<<wb, 256, 0, stream>>>(Fp_w, wFp, wn);
  w2bf_kernel<<<wb, 256, 0, stream>>>(RW_w, wRW, wn);

  // 3) projections (all WMMA GEMMs)
  proj_on_kernel<<<dim3(Ci_ / 16, N_ / 256, B_), 128, 0, stream>>>(snT, wRg, Rg_b, g);
  proj_nt_kernel<<<dim3(N_ / 16, B_), 128, 0, stream>>>(snT, wRt, Rt_b, sth);
  proj_nt_kernel<<<dim3(N_ / 16, B_), 128, 0, stream>>>(snT, wRp, Rp_b, sph);
  proj_nt_kernel<<<dim3(N_ / 16, B_), 128, 0, stream>>>(mnT, wFt, Ft_b, th);
  proj_nt_kernel<<<dim3(N_ / 16, B_), 128, 0, stream>>>(mnT, wFp, Fp_b, ph);

  // 4) fused dual-score flash attention with TDM double-buffered key tiles
  flash_attn_kernel<<<dim3(N_ / 64, B_), 128, 0, stream>>>(th, ph, sth, sph, g,
                                                           alpha, ybf);

  // 5) RW 1x1 conv + bias + residual
  rw_residual_kernel<<<dim3(C_ / 16, N_ / 256, B_), 128, 0, stream>>>(
      ybf, wRW, RW_b, self_fea, (float*)d_out);
}
